// T5SelfAttention_6880537608801
// MI455X (gfx1250) — matrix operand order repaired and checked by reading, not compiled, not run on hardware
//
#include <hip/hip_runtime.h>
#include <hip/hip_bf16.h>

// ---------------------------------------------------------------------------
// T5 self-attention forward for MI455X (gfx1250).
// All GEMM stages via v_wmma_f32_16x16x32_f16 (wave32 WMMA), f32 softmax.
// - GEMMs: 16x64 tiles/wave, reused fragment, packed 16B stores.
// - QK^T computed transposed (M=key) so S stores to LDS are ds_store_b128.
// - Bias row staged to LDS with global_load_async_to_lds_b128 (ASYNCcnt),
//   overlapped with phase-1 WMMA.
// ---------------------------------------------------------------------------

typedef _Float16 half_t;
typedef __attribute__((ext_vector_type(8)))  _Float16 v8h;
typedef __attribute__((ext_vector_type(16))) _Float16 v16h;
typedef __attribute__((ext_vector_type(8)))  float    v8f;
typedef __attribute__((ext_vector_type(4)))  float    v4f;

#define BB   2
#define NN   2048
#define DIMD 768
#define HH   12
#define DHD  64
#define MROWS (BB*NN)        // 4096
#define SCALE 0.125f         // DH^-0.5
#define SROW_STRIDE 2056     // padded LDS row stride (f16 elems), 16B-aligned
#define SP_BYTES (32 * SROW_STRIDE * 2)          // 131584
#define L_BYTES  128
#define BIAS_OFF (SP_BYTES + L_BYTES)            // bias row after l[]
#define SMEM_BYTES (BIAS_OFF + NN * 4)           // +8192 -> 139904

__device__ __forceinline__ v16h ld2x8(const half_t* p0, const half_t* p1) {
    v8h lo = *(const v8h*)p0;
    v8h hi = *(const v8h*)p1;
    return __builtin_shufflevector(lo, hi, 0,1,2,3,4,5,6,7,8,9,10,11,12,13,14,15);
}

__device__ __forceinline__ v8f wmma16(v16h a, v16h b, v8f c) {
    // D = A(16x32) * B(32x16) + C, f32 accumulate
    return __builtin_amdgcn_wmma_f32_16x16x32_f16(false, a, false, b,
                                                  (short)0, c, false, false);
}

#define V8F_ZERO {0.f,0.f,0.f,0.f,0.f,0.f,0.f,0.f}

// ---------------------------------------------------------------------------
// Prep kernels
// ---------------------------------------------------------------------------
__global__ void k_cvt_f32_to_f16(const float* __restrict__ in,
                                 half_t* __restrict__ out, int n) {
    int i = blockIdx.x * blockDim.x + threadIdx.x;
    if (i < n) out[i] = (half_t)in[i];
}

// Wt[c*768 + k] = (f16) W[k*768 + c]
__global__ void k_transpose_w(const float* __restrict__ W,
                              half_t* __restrict__ Wt) {
    int o = blockIdx.x * blockDim.x + threadIdx.x;
    if (o < DIMD * DIMD) {
        int c = o / DIMD, k = o % DIMD;
        Wt[o] = (half_t)W[k * DIMD + c];
    }
}

// bias_tab[h*2048 + delta] = rel_emb[bucket(delta)*H + h] * scale
__global__ void k_build_bias(const float* __restrict__ rel_emb,
                             float* __restrict__ tab) {
    int o = blockIdx.x * blockDim.x + threadIdx.x;
    if (o < HH * NN) {
        int h = o / NN, d = o % NN;
        int bucket;
        if (d < 16) {
            bucket = d;
        } else {
            int v = 16 + (int)(logf((float)d * (1.0f/16.0f)) * (16.0f / logf(8.0f)));
            bucket = v < 31 ? v : 31;
        }
        tab[o] = rel_emb[bucket * HH + h] * SCALE;
    }
}

// ---------------------------------------------------------------------------
// WMMA GEMM: C[M x 768] = alpha * A[M x 768] * Bt[768 x 768]^T
//   A row-major f16; Bt = transposed weights (column-contiguous) f16.
//   MODE 0: Ch row-major f16          (q, k projections)  [transposed compute]
//   MODE 1: Ch = v_t[b][col][seq] f16 (V projection)      [direct compute]
//   MODE 2: Cf row-major f32          (output projection) [transposed compute]
// 16(M) x 64(N) tile per wave: 4 accumulators, the shared fragment (x row or
// x row) is loaded once per K-step and reused across 4 WMMAs.
// For MODE 0/2 we compute D^T (A=weight cols, B=x rows) so that each lane's
// 8 accumulator values are contiguous along the output row -> packed stores.
// ---------------------------------------------------------------------------
template <int MODE>
__global__ __launch_bounds__(256)
void k_gemm_wmma(const half_t* __restrict__ A, const half_t* __restrict__ Bt,
                 half_t* __restrict__ Ch, float* __restrict__ Cf, float alpha) {
    const int wave = threadIdx.x >> 5;
    const int lane = threadIdx.x & 31;
    const int tile = blockIdx.x * 8 + wave;
    const int tilesN64 = DIMD / 64;                  // 12
    if (tile >= (MROWS / 16) * tilesN64) return;
    const int m0 = (tile / tilesN64) * 16;
    const int n0 = (tile % tilesN64) * 64;
    const int c  = lane & 15;
    const int g  = lane >> 4;

    const half_t* arow = A  + (size_t)(m0 + c) * DIMD;   // x row m0+c
    const half_t* bcol = Bt + (size_t)(n0 + c) * DIMD;   // weight col n0+c (+t*16*768)

    v8f acc0 = V8F_ZERO, acc1 = V8F_ZERO, acc2 = V8F_ZERO, acc3 = V8F_ZERO;

    #pragma unroll 2
    for (int k0 = 0; k0 < DIMD; k0 += 32) {
        __builtin_prefetch(arow + k0 + 64, 0, 3);
        __builtin_prefetch(bcol + k0 + 64, 0, 3);
        if (MODE == 1) {
            // direct: D(M=x row, N=col). A-frag = x row, B-frags = weight cols.
            v16h a  = ld2x8(arow + k0 + 16*g, arow + k0 + 16*g + 8);
            v16h b0 = ld2x8(bcol + k0 + 16*g,           bcol + k0 + 16*g + 8);
            v16h b1 = ld2x8(bcol + 16*DIMD + k0 + 16*g, bcol + 16*DIMD + k0 + 16*g + 8);
            v16h b2 = ld2x8(bcol + 32*DIMD + k0 + 16*g, bcol + 32*DIMD + k0 + 16*g + 8);
            v16h b3 = ld2x8(bcol + 48*DIMD + k0 + 16*g, bcol + 48*DIMD + k0 + 16*g + 8);
            acc0 = wmma16(a, b0, acc0);
            acc1 = wmma16(a, b1, acc1);
            acc2 = wmma16(a, b2, acc2);
            acc3 = wmma16(a, b3, acc3);
        } else {
            // transposed: D(M=col, N=x row). A-frags = weight cols, B-frag = x row.
            v16h b  = ld2x8(arow + k0 + 8*g, arow + k0 + 16 + 8*g);
            v16h a0 = ld2x8(bcol + k0 + 8*g,            bcol + k0 + 16 + 8*g);
            v16h a1 = ld2x8(bcol + 16*DIMD + k0 + 8*g,  bcol + 16*DIMD + k0 + 16 + 8*g);
            v16h a2 = ld2x8(bcol + 32*DIMD + k0 + 8*g,  bcol + 32*DIMD + k0 + 16 + 8*g);
            v16h a3 = ld2x8(bcol + 48*DIMD + k0 + 8*g,  bcol + 48*DIMD + k0 + 16 + 8*g);
            acc0 = wmma16(a0, b, acc0);
            acc1 = wmma16(a1, b, acc1);
            acc2 = wmma16(a2, b, acc2);
            acc3 = wmma16(a3, b, acc3);
        }
    }

    v8f accs[4] = {acc0, acc1, acc2, acc3};
    #pragma unroll
    for (int t = 0; t < 4; ++t) {
        if (MODE == 0) {
            // element (col = n0+t*16+8g+r, row = m0+c): contiguous along r.
            v8h pk;
            #pragma unroll
            for (int r = 0; r < 8; ++r) pk[r] = (half_t)(accs[t][r] * alpha);
            *(v8h*)(Ch + (size_t)(m0 + c) * DIMD + n0 + t*16 + 8*g) = pk;
        } else if (MODE == 1) {
            // v_t[b][col][seq]; lane: col = n0+t*16+c, seq = m0%2048 + 8g + r.
            const int bb = m0 >> 11;
            const int j0 = (m0 & 2047) + 8*g;
            v8h pk;
            #pragma unroll
            for (int r = 0; r < 8; ++r) pk[r] = (half_t)(accs[t][r] * alpha);
            *(v8h*)(Ch + (size_t)bb * DIMD * NN + (size_t)(n0 + t*16 + c) * NN + j0) = pk;
        } else {
            v4f lo = {accs[t][0]*alpha, accs[t][1]*alpha, accs[t][2]*alpha, accs[t][3]*alpha};
            v4f hi = {accs[t][4]*alpha, accs[t][5]*alpha, accs[t][6]*alpha, accs[t][7]*alpha};
            float* p = Cf + (size_t)(m0 + c) * DIMD + n0 + t*16 + 8*g;
            *(v4f*)p       = lo;
            *(v4f*)(p + 4) = hi;
        }
    }
}

// ---------------------------------------------------------------------------
// Fused attention: per block = one (b, h, 32-query tile), 8 waves.
// LDS: S/P [32][2056] f16 + l[32] f32 + bias_row[2048] f32 (139904 B dynamic;
// CDNA5 WGP has 320KB LDS).
// ---------------------------------------------------------------------------
__global__ __launch_bounds__(256)
void k_attn_wmma(const half_t* __restrict__ qh,   // [B,N,768] f16 (pre-scaled)
                 const half_t* __restrict__ kh,   // [B,N,768] f16
                 const half_t* __restrict__ vt,   // [B,768,N] f16 transposed
                 const float*  __restrict__ bias_tab, // [H,2048] (pre-scaled)
                 const unsigned char* __restrict__ mask, // [B,N]
                 half_t* __restrict__ att) {      // [B,N,768] f16
    extern __shared__ char smem_raw[];
    half_t* sp       = (half_t*)smem_raw;                  // [32][2056]
    float*  l_lds    = (float*)(smem_raw + SP_BYTES);      // [32]
    float*  bias_lds = (float*)(smem_raw + BIAS_OFF);      // [2048]

    const int qblock = blockIdx.x;       // 0..63 (32 queries each)
    const int h      = blockIdx.y;
    const int bb     = blockIdx.z;
    const int wave   = threadIdx.x >> 5;
    const int lane   = threadIdx.x & 31;
    const int c      = lane & 15;
    const int g      = lane >> 4;
    const int qt     = wave >> 2;        // which 16-query sub-tile (0/1)
    const int i0     = qblock * 32 + qt * 16;

    // ---- Kick off async global->LDS copy of this head's bias row (8KB).
    //      Tracked by ASYNCcnt; overlaps with phase-1 WMMA below. ----
    {
        const float* bsrc = bias_tab + (size_t)h * NN;
        #pragma unroll
        for (int it = 0; it < 2; ++it) {
            const int fidx = (it * 256 + (int)threadIdx.x) * 4;     // float index
            const unsigned lds_addr = (unsigned)BIAS_OFF + (unsigned)fidx * 4u;
            const unsigned long long gaddr =
                (unsigned long long)(const void*)(bsrc + fidx);
            asm volatile("global_load_async_to_lds_b128 %0, %1, off"
                         :: "v"(lds_addr), "v"(gaddr) : "memory");
        }
    }

    // ---- Phase 1: S^T tiles = K * Q^T  (D: M=key, N=query) ----
    // This orientation makes each lane's 8 outputs contiguous along the key
    // axis of the [query][key] LDS layout -> one ds_store_b128 per fragment.
    {
        const half_t* qcol = qh + ((size_t)(bb * NN + i0 + c)) * DIMD + h * DHD;
        // Q fragments (B operand) are invariant over key tiles: load once.
        const v16h bq0 = ld2x8(qcol + 16*g,      qcol + 16*g + 8);       // d 0..31
        const v16h bq1 = ld2x8(qcol + 32 + 16*g, qcol + 32 + 16*g + 8);  // d 32..63
        for (int kt = wave & 3; kt * 16 < i0 + 16; kt += 4) {
            const int j0 = kt * 16;
            const half_t* krow = kh + ((size_t)(bb * NN + j0 + c)) * DIMD + h * DHD;
            v16h a0 = ld2x8(krow + 16*g,      krow + 16*g + 8);          // d 0..31
            v16h a1 = ld2x8(krow + 32 + 16*g, krow + 32 + 16*g + 8);     // d 32..63
            v8f acc = V8F_ZERO;
            acc = wmma16(a0, bq0, acc);
            acc = wmma16(a1, bq1, acc);
            // lane: query = qt*16 + c, keys = j0 + 8g + r (contiguous)
            v8h pk;
            #pragma unroll
            for (int r = 0; r < 8; ++r) pk[r] = (half_t)acc[r];
            *(v8h*)(sp + (size_t)(qt*16 + c) * SROW_STRIDE + j0 + 8*g) = pk;
        }
    }
    // Bias row must be resident before softmax.
    asm volatile("s_wait_asynccnt 0x0" ::: "memory");
    __syncthreads();

    // ---- Phase 2/3: softmax (bias + causal), write P f16 in place ----
    {
        #pragma unroll
        for (int rr = 0; rr < 4; ++rr) {
            const int row   = wave * 4 + rr;                 // 0..31
            const int i     = qblock * 32 + row;             // global query
            const int jmax  = qblock * 32 + ((row >> 4) + 1) * 16; // tile end
            const int jfill = (jmax + 31) & ~31;             // pad to K=32
            const bool rowvalid = mask[bb * NN + i] != 0;
            half_t* srow = sp + row * SROW_STRIDE;

            float m = -3.0e38f;
            for (int j = lane; j <= i; j += 32) {
                float s = rowvalid ? ((float)srow[j] + bias_lds[i - j]) : 0.0f;
                m = fmaxf(m, s);
            }
            for (int off = 16; off; off >>= 1)
                m = fmaxf(m, __shfl_xor(m, off, 32));

            float sum = 0.0f;
            for (int j = lane; j < jfill; j += 32) {
                float p = 0.0f;
                if (j <= i) {
                    float s = rowvalid ? ((float)srow[j] + bias_lds[i - j]) : 0.0f;
                    p = __expf(s - m);
                    sum += p;
                }
                srow[j] = (half_t)p;
            }
            for (int off = 16; off; off >>= 1)
                sum += __shfl_xor(sum, off, 32);
            if (lane == 0) l_lds[row] = sum;
        }
    }
    __syncthreads();

    // ---- Phase 4: O = P * V, normalize, store f16 ----
    {
        const int dt   = wave & 3;                           // 16-wide d tile
        const int jend = i0 + 16;                            // causal K extent
        const half_t* vcol =
            vt + (size_t)bb * DIMD * NN + (size_t)(h * DHD + dt * 16 + c) * NN;
        v8f acc = V8F_ZERO;
        for (int j0 = 0; j0 < jend; j0 += 32) {
            __builtin_prefetch(vcol + j0 + 64, 0, 3);
            const half_t* prow = sp + (size_t)(qt * 16 + c) * SROW_STRIDE + j0;
            v16h a = ld2x8(prow + 16*g, prow + 16*g + 8);    // P from LDS
            v16h b = ld2x8(vcol + j0 + 16*g, vcol + j0 + 16*g + 8);
            acc = wmma16(a, b, acc);
        }
        #pragma unroll
        for (int r = 0; r < 8; ++r) {
            const int row = qt * 16 + 8 * g + r;
            const float o = acc[r] / l_lds[row];
            const int i   = qblock * 32 + row;
            att[((size_t)(bb * NN + i)) * DIMD + h * DHD + dt * 16 + c] = (half_t)o;
        }
    }
}

// ---------------------------------------------------------------------------
// Host launcher
// ---------------------------------------------------------------------------
extern "C" void kernel_launch(void* const* d_in, const int* in_sizes, int n_in,
                              void* d_out, int out_size, void* d_ws, size_t ws_size,
                              hipStream_t stream) {
    (void)in_sizes; (void)n_in; (void)out_size; (void)ws_size;
    const float* x       = (const float*)d_in[0];
    const unsigned char* mask = (const unsigned char*)d_in[1];
    const float* Wq      = (const float*)d_in[2];
    const float* Wk      = (const float*)d_in[3];
    const float* Wv      = (const float*)d_in[4];
    const float* Wo      = (const float*)d_in[5];
    const float* rel_emb = (const float*)d_in[6];
    float* out = (float*)d_out;

    // workspace partition (256B aligned chunks)
    char* ws = (char*)d_ws;
    size_t off = 0;
    auto alloc = [&](size_t bytes) -> char* {
        char* p = ws + off;
        off += (bytes + 255) & ~(size_t)255;
        return p;
    };
    half_t* x_h   = (half_t*)alloc((size_t)MROWS * DIMD * 2);
    half_t* wq_t  = (half_t*)alloc((size_t)DIMD * DIMD * 2);
    half_t* wk_t  = (half_t*)alloc((size_t)DIMD * DIMD * 2);
    half_t* wv_t  = (half_t*)alloc((size_t)DIMD * DIMD * 2);
    half_t* wo_t  = (half_t*)alloc((size_t)DIMD * DIMD * 2);
    half_t* q_h   = (half_t*)alloc((size_t)MROWS * DIMD * 2);
    half_t* k_h   = (half_t*)alloc((size_t)MROWS * DIMD * 2);
    half_t* v_t   = (half_t*)alloc((size_t)MROWS * DIMD * 2);
    half_t* att_h = (half_t*)alloc((size_t)MROWS * DIMD * 2);
    float*  btab  = (float*) alloc((size_t)HH * NN * 4);

    // --- prep ---
    {
        int n = MROWS * DIMD;
        k_cvt_f32_to_f16<<<(n + 255) / 256, 256, 0, stream>>>(x, x_h, n);
        int nw = DIMD * DIMD;
        k_transpose_w<<<(nw + 255) / 256, 256, 0, stream>>>(Wq, wq_t);
        k_transpose_w<<<(nw + 255) / 256, 256, 0, stream>>>(Wk, wk_t);
        k_transpose_w<<<(nw + 255) / 256, 256, 0, stream>>>(Wv, wv_t);
        k_transpose_w<<<(nw + 255) / 256, 256, 0, stream>>>(Wo, wo_t);
        int nb = HH * NN;
        k_build_bias<<<(nb + 255) / 256, 256, 0, stream>>>(rel_emb, btab);
    }

    // --- projections (WMMA, 16x64 tiles per wave) ---
    const int gemm_blocks = (MROWS / 16) * (DIMD / 64) / 8;  // 384
    k_gemm_wmma<0><<<gemm_blocks, 256, 0, stream>>>(x_h, wq_t, q_h, nullptr, SCALE);
    k_gemm_wmma<0><<<gemm_blocks, 256, 0, stream>>>(x_h, wk_t, k_h, nullptr, 1.0f);
    k_gemm_wmma<1><<<gemm_blocks, 256, 0, stream>>>(x_h, wv_t, v_t, nullptr, 1.0f);

    // --- fused attention (WMMA + softmax + async LDS staging) ---
    k_attn_wmma<<<dim3(NN / 32, HH, BB), 256, SMEM_BYTES, stream>>>(
        q_h, k_h, v_t, btab, mask, att_h);

    // --- output projection (WMMA, f32 out) ---
    k_gemm_wmma<2><<<gemm_blocks, 256, 0, stream>>>(att_h, wo_t, nullptr, out, 1.0f);
}
